// CausalSelfAttention_3959959847298
// MI455X (gfx1250) — compile-verified, matrix operand order checked by reading, not run on hardware
//
#include <hip/hip_runtime.h>
#include <hip/hip_bf16.h>

typedef __attribute__((ext_vector_type(16))) _Float16 v16h;
typedef __attribute__((ext_vector_type(8)))  float    v8f;
typedef __attribute__((ext_vector_type(4)))  int      v4i;

#define Bs 4
#define Ts 2048
#define Cs 1024
#define Hs 16
#define Ds 64
#define BT (Bs*Ts)          // 8192 rows

#define GLOBAL_AS __attribute__((address_space(1)))
#define LDS_AS    __attribute__((address_space(3)))

__device__ inline v8f wmma16(v16h a, v16h b, v8f c) {
  return __builtin_amdgcn_wmma_f32_16x16x32_f16(false, a, false, b, (short)0, c, false, false);
}

// 16-byte global -> LDS copy; async (ASYNCcnt-tracked) when the CDNA5 builtin exists.
__device__ inline void async_copy16(const _Float16* g, _Float16* l) {
#if defined(__has_builtin) && __has_builtin(__builtin_amdgcn_global_load_async_to_lds_b128)
  __builtin_amdgcn_global_load_async_to_lds_b128(
      (GLOBAL_AS v4i*)(g), (LDS_AS v4i*)(l), 0, 0);
#else
  *(uint4*)l = *(const uint4*)g;    // synchronous fallback
#endif
}

template <int N>
__device__ inline void wait_async() {
#if defined(__has_builtin) && __has_builtin(__builtin_amdgcn_s_wait_asynccnt)
  __builtin_amdgcn_s_wait_asynccnt((unsigned short)N);
#else
  asm volatile("s_wait_asynccnt %0" ::"i"(N) : "memory");
#endif
}

// Load a 16x32 fragment (A layout; identical for B when source is [N][K] row-major).
// lane&15 -> row (M or N), halves: K = (lane>>4)*8 + i + (i&8)
template <typename PT>
__device__ inline v16h load_rm(const PT* src, int ld) {
  const int lane = threadIdx.x & 31;
  const PT* p = src + (lane & 15) * ld + (lane >> 4) * 8;
  v16h f;
#pragma unroll
  for (int i = 0; i < 16; i += 2) {
    int kb = i + (i & 8);
    f[i]     = (_Float16)p[kb];
    f[i + 1] = (_Float16)p[kb + 1];
  }
  return f;
}

// Load a B fragment from a [K][N] row-major source (stride along K).
template <typename PT>
__device__ inline v16h load_cm(const PT* src, int ld) {
  const int lane = threadIdx.x & 31;
  const PT* p = src + (lane & 15) + (lane >> 4) * 8 * ld;
  v16h f;
#pragma unroll
  for (int i = 0; i < 16; ++i) {
    int kk = i + (i & 8);
    f[i] = p[kk * ld];
  }
  return f;
}

__global__ void cvt_f32_f16(const float* __restrict__ src, _Float16* __restrict__ dst, int n) {
  int i = blockIdx.x * blockDim.x + threadIdx.x;
  if (i < n) dst[i] = (_Float16)src[i];
}

// ---------------------------------------------------------------------------
// LDS-staged GEMM core: block = 256 threads (8 waves), block tile 128(M)x64(N),
// K stepped by 32 with double-buffered async global->LDS staging.
// A: [M][K] row-major (f16). B: [N][K] row-major (f16).  acc = A @ B^T
// ---------------------------------------------------------------------------
__device__ inline void stage_tiles(const _Float16* __restrict__ A,
                                   const _Float16* __restrict__ Bm,
                                   int m0, int n0, int kk,
                                   _Float16* xs, _Float16* bs) {
  const int tid = threadIdx.x;
  // A tile: 128 rows x 32 halves = 512 chunks of 16B; 2 per thread
#pragma unroll
  for (int rep = 0; rep < 2; ++rep) {
    int c   = tid + rep * 256;
    int row = c >> 2, kc = (c & 3) * 8;
    async_copy16(A + (size_t)(m0 + row) * Cs + kk + kc, xs + row * 32 + kc);
  }
  // B tile: 64 rows x 32 halves = 256 chunks of 16B; 1 per thread
  {
    int row = tid >> 2, kc = (tid & 3) * 8;
    async_copy16(Bm + (size_t)(n0 + row) * Cs + kk + kc, bs + row * 32 + kc);
  }
}

template <typename EPI>
__device__ inline void gemm_block(const _Float16* __restrict__ A,
                                  const _Float16* __restrict__ Bm,
                                  int m0, int n0, EPI epilogue) {
  __shared__ _Float16 xs[2][128 * 32];   // 16 KB
  __shared__ _Float16 bsl[2][64 * 32];   //  8 KB
  const int wv = threadIdx.x >> 5;
  const int KSTEPS = Cs / 32;            // 32

  v8f acc[4] = {};
  stage_tiles(A, Bm, m0, n0, 0, xs[0], bsl[0]);
  for (int step = 0; step < KSTEPS; ++step) {
    const int buf = step & 1;
    if (step + 1 < KSTEPS) {
      stage_tiles(A, Bm, m0, n0, (step + 1) * 32, xs[buf ^ 1], bsl[buf ^ 1]);
      wait_async<3>();                   // current buffer's 3 ops retired (in-order)
    } else {
      wait_async<0>();
    }
    __syncthreads();
    v16h a = load_rm(&xs[buf][(wv * 16) * 32], 32);
#pragma unroll
    for (int j = 0; j < 4; ++j) {
      v16h b = load_rm(&bsl[buf][(j * 16) * 32], 32);
      acc[j] = wmma16(a, b, acc[j]);
    }
    __syncthreads();
  }
  epilogue(acc, m0 + wv * 16, n0);
}

// qkv = x @ w_attn^T + b_attn, scattered to q/k/v as [B,H,T,D] f16
__global__ void __launch_bounds__(256)
gemm_qkv(const _Float16* __restrict__ X, const _Float16* __restrict__ W,
         const float* __restrict__ bias,
         _Float16* __restrict__ q, _Float16* __restrict__ k, _Float16* __restrict__ v) {
  const int n0 = blockIdx.x * 64;
  const int m0 = blockIdx.y * 128;
  const int lane = threadIdx.x & 31;
  const int grp = lane >> 4, nn = lane & 15;
  gemm_block(X, W, m0, n0, [&](v8f (&acc)[4], int mw0, int nb0) {
    const int which = nb0 / Cs;          // whole 64-tile stays in one of q/k/v
    _Float16* dst = which == 0 ? q : (which == 1 ? k : v);
#pragma unroll
    for (int j = 0; j < 4; ++j) {
      int n = nb0 + j * 16 + nn;
      int c = n % Cs;
      int h = c >> 6, d = c & 63;
      float bb = bias[n];
#pragma unroll
      for (int r = 0; r < 8; ++r) {
        int m  = mw0 + r + 8 * grp;      // token row in [0, BT)
        int b_ = m >> 11, t = m & (Ts - 1);
        dst[(((size_t)(b_ * Hs + h) * Ts + t) << 6) + d] = (_Float16)(acc[j][r] + bb);
      }
    }
  });
}

// out = y @ w_proj^T + b_proj  (f32 output)
__global__ void __launch_bounds__(256)
gemm_proj(const _Float16* __restrict__ Yh, const _Float16* __restrict__ W,
          const float* __restrict__ bias, float* __restrict__ out) {
  const int n0 = blockIdx.x * 64;
  const int m0 = blockIdx.y * 128;
  const int lane = threadIdx.x & 31;
  const int grp = lane >> 4, nn = lane & 15;
  gemm_block(Yh, W, m0, n0, [&](v8f (&acc)[4], int mw0, int nb0) {
#pragma unroll
    for (int j = 0; j < 4; ++j) {
      int n = nb0 + j * 16 + nn;
      float bb = bias[n];
#pragma unroll
      for (int r = 0; r < 8; ++r) {
        int m = mw0 + r + 8 * grp;
        out[(size_t)m * Cs + n] = acc[j][r] + bb;
      }
    }
  });
}

// ---------------------------------------------------------------------------
// Flash-style causal attention: one wave per (b,h, 16-query tile).
// V tiles are async-staged to LDS overlapping the QK^T WMMAs + softmax.
// ---------------------------------------------------------------------------
__global__ void __launch_bounds__(32)
attn_fwd(const _Float16* __restrict__ Q, const _Float16* __restrict__ K,
         const _Float16* __restrict__ V, _Float16* __restrict__ Y) {
  __shared__ _Float16 plds[16 * 32];     // P tile (1 KB)
  __shared__ _Float16 vlds[32 * 64];     // V tile (4 KB)
  const int lane = threadIdx.x & 31;
  const int grp = lane >> 4, nn = lane & 15;
  const int idx = blockIdx.x;            // B*H * (T/16)
  const int bh = idx >> 7;               // T/16 = 128
  const int qt = idx & 127;
  const int q0 = qt * 16;
  const _Float16* qp = Q + (size_t)bh * Ts * Ds;
  const _Float16* kp = K + (size_t)bh * Ts * Ds;
  const _Float16* vp = V + (size_t)bh * Ts * Ds;

  v16h aq0 = load_rm(qp + q0 * Ds, Ds);
  v16h aq1 = load_rm(qp + q0 * Ds + 32, Ds);

  v8f acc[4] = {};
  float mrun[8], lrun[8];
#pragma unroll
  for (int r = 0; r < 8; ++r) { mrun[r] = -3.0e38f; lrun[r] = 0.0f; }

  const float sc = 0.125f;               // 1/sqrt(64)
  const int nkb = (q0 + 47) >> 5;        // causal bound on 32-key blocks

  for (int kb = 0; kb < nkb; ++kb) {
    const int kbase = kb * 32;

    // async stage V tile (32x64 halves = 4 KB = 256 x 16B chunks; 8 per lane)
    const _Float16* vsrc = vp + (size_t)kbase * Ds;
#pragma unroll
    for (int c = 0; c < 8; ++c) {
      int off = (c * 32 + lane) * 8;     // halves
      async_copy16(vsrc + off, vlds + off);
    }

    v8f s[2];
#pragma unroll
    for (int t = 0; t < 2; ++t) {
      int n0 = kbase + t * 16;
      v16h bk0 = load_rm(kp + (size_t)n0 * Ds, Ds);
      v16h bk1 = load_rm(kp + (size_t)n0 * Ds + 32, Ds);
      v8f z = {};
      z = wmma16(aq0, bk0, z);
      z = wmma16(aq1, bk1, z);
      int n = n0 + nn;
#pragma unroll
      for (int r = 0; r < 8; ++r) {
        int m = q0 + r + 8 * grp;
        float val = z[r] * sc;
        s[t][r] = (n <= m) ? val : -3.0e38f;   // causal mask
      }
    }
    float corr[8];
#pragma unroll
    for (int r = 0; r < 8; ++r) {
      float mx = fmaxf(s[0][r], s[1][r]);
      mx = fmaxf(mx, __shfl_xor(mx, 1, 32));
      mx = fmaxf(mx, __shfl_xor(mx, 2, 32));
      mx = fmaxf(mx, __shfl_xor(mx, 4, 32));
      mx = fmaxf(mx, __shfl_xor(mx, 8, 32));
      float mnew = fmaxf(mrun[r], mx);
      float p0 = __expf(s[0][r] - mnew);
      float p1 = __expf(s[1][r] - mnew);
      s[0][r] = p0; s[1][r] = p1;
      float ps = p0 + p1;
      ps += __shfl_xor(ps, 1, 32);
      ps += __shfl_xor(ps, 2, 32);
      ps += __shfl_xor(ps, 4, 32);
      ps += __shfl_xor(ps, 8, 32);
      float cf = __expf(mrun[r] - mnew);
      lrun[r] = lrun[r] * cf + ps;
      mrun[r] = mnew;
      corr[r] = cf;
    }
#pragma unroll
    for (int j = 0; j < 4; ++j)
#pragma unroll
      for (int r = 0; r < 8; ++r) acc[j][r] *= corr[r];

    // pack P (16x32 f16) to LDS, reload in A-fragment layout
#pragma unroll
    for (int t = 0; t < 2; ++t)
#pragma unroll
      for (int r = 0; r < 8; ++r)
        plds[(r + 8 * grp) * 32 + t * 16 + nn] = (_Float16)s[t][r];
    asm volatile("s_wait_dscnt 0" ::: "memory");   // P store -> reload (same wave)
    wait_async<0>();                               // V tile landed in LDS
    v16h pa = load_rm(plds, 32);
#pragma unroll
    for (int j = 0; j < 4; ++j) {
      v16h bv = load_cm(vlds + j * 16, Ds);
      acc[j] = wmma16(pa, bv, acc[j]);
    }
  }

  // epilogue: y[b, t, h*64 + d] = acc / l
  const int b_ = bh >> 4, h = bh & 15;
#pragma unroll
  for (int j = 0; j < 4; ++j) {
    int d = j * 16 + nn;
#pragma unroll
    for (int r = 0; r < 8; ++r) {
      int tok = q0 + r + 8 * grp;
      float inv = 1.0f / lrun[r];
      Y[(size_t)(b_ * Ts + tok) * Cs + h * Ds + d] = (_Float16)(acc[j][r] * inv);
    }
  }
}

extern "C" void kernel_launch(void* const* d_in, const int* in_sizes, int n_in,
                              void* d_out, int out_size, void* d_ws, size_t ws_size,
                              hipStream_t stream) {
  const float* x      = (const float*)d_in[0];
  const float* w_attn = (const float*)d_in[1];
  const float* b_attn = (const float*)d_in[2];
  const float* w_proj = (const float*)d_in[3];
  const float* b_proj = (const float*)d_in[4];
  float* out = (float*)d_out;

  // workspace layout (f16 elements)
  _Float16* base = (_Float16*)d_ws;
  const size_t nX  = (size_t)BT * Cs;           // 8,388,608
  const size_t nWA = (size_t)3 * Cs * Cs;       // 3,145,728
  const size_t nWP = (size_t)Cs * Cs;           // 1,048,576
  const size_t nH  = (size_t)Bs * Hs * Ts * Ds; // 8,388,608
  _Float16* xh  = base;
  _Float16* wAh = xh  + nX;
  _Float16* wPh = wAh + nWA;
  _Float16* qh  = wPh + nWP;
  _Float16* kh  = qh  + nH;
  _Float16* vh  = kh  + nH;
  _Float16* yh  = vh  + nH;

  cvt_f32_f16<<<(int)((nX  + 255) / 256), 256, 0, stream>>>(x,      xh,  (int)nX);
  cvt_f32_f16<<<(int)((nWA + 255) / 256), 256, 0, stream>>>(w_attn, wAh, (int)nWA);
  cvt_f32_f16<<<(int)((nWP + 255) / 256), 256, 0, stream>>>(w_proj, wPh, (int)nWP);

  // QKV: block tile 128x64 -> grid (3072/64, 8192/128)
  gemm_qkv<<<dim3(48, 64), 256, 0, stream>>>(xh, wAh, b_attn, qh, kh, vh);

  // Attention: B*H * (T/16) = 64*128 = 8192 single-wave blocks
  attn_fwd<<<8192, 32, 0, stream>>>(qh, kh, vh, yh);

  // Projection: block tile 128x64 -> grid (1024/64, 8192/128)
  gemm_proj<<<dim3(16, 64), 256, 0, stream>>>(yh, wPh, b_proj, out);
}
